// LCSA_5497558139713
// MI455X (gfx1250) — matrix operand order repaired and checked by reading, not compiled, not run on hardware
//
#include <hip/hip_runtime.h>
#include <math.h>

// LCSA: dilated local-context self-attention, bf16 WMMA pipeline for gfx1250.
// B=2 S=2048 D=1024 H=8 E=128 KW=16

#define BB  2
#define SS  2048
#define DD  1024
#define HH  8
#define EE  128
#define KWW 16

typedef __attribute__((ext_vector_type(16))) __bf16 v16bf;
typedef __attribute__((ext_vector_type(8)))  __bf16 v8bf;
typedef __attribute__((ext_vector_type(8)))  float  v8f;
typedef __attribute__((ext_vector_type(4)))  unsigned int u32x4;
typedef __attribute__((ext_vector_type(8)))  unsigned int u32x8;

typedef int v4i __attribute__((vector_size(16)));
typedef __attribute__((address_space(1))) v4i GV4i;  // global 16B vector
typedef __attribute__((address_space(3))) v4i LV4i;  // LDS 16B vector
typedef __attribute__((address_space(3))) void LAS;  // LDS void

#if __has_builtin(__builtin_amdgcn_global_load_async_to_lds_b128) && \
    __has_builtin(__builtin_amdgcn_s_wait_asynccnt)
#define HAS_ASYNC_LDS 1
#else
#define HAS_ASYNC_LDS 0
#endif

static __device__ inline v16bf cat8(v8bf a, v8bf b) {
  return __builtin_shufflevector(a, b, 0,1,2,3,4,5,6,7,8,9,10,11,12,13,14,15);
}
static __device__ inline v8f wmma_bf16(v16bf a, v16bf b, v8f c) {
  // D = A(16x32 bf16) * B(32x16 bf16) + C(16x16 f32)
  return __builtin_amdgcn_wmma_f32_16x16x32_bf16(false, a, false, b, (short)0, c,
                                                 false, false);
}

// ---------------------------------------------------------------- conversions
__global__ void cvt_f32_bf16(const float* __restrict__ s, __bf16* __restrict__ d,
                             int n) {
  int i = blockIdx.x * blockDim.x + threadIdx.x;
  if (i < n) d[i] = (__bf16)s[i];
}

// ------------------------------------------------------------ QKV projection
// C[4096x128] = X[4096x1024] * W[1024x128] + bias; one (proj, head) per blockIdx.y
// A tile staged via Tensor Data Mover (tensor_load_to_lds); B via LDS transpose.
__global__ __launch_bounds__(256) void qkv_gemm(
    const __bf16* __restrict__ X,
    const __bf16* __restrict__ Wq, const __bf16* __restrict__ Wk,
    const __bf16* __restrict__ Wv,
    const float* __restrict__ bq, const float* __restrict__ bk,
    const float* __restrict__ bv,
    __bf16* __restrict__ Q, __bf16* __restrict__ K, __bf16* __restrict__ V) {
  __shared__ __align__(16) __bf16 As[128][40];  // [m][k], +8 pad (TDM pads)
  __shared__ __align__(16) __bf16 Bs[128][40];  // [n][k] (transposed), +8 pad
  const int tid = threadIdx.x, lane = tid & 31, w = tid >> 5;
  const int wm = w & 1, wn = w >> 1;            // 2 waves along M, 4 along N
  const int m0 = blockIdx.x * 128;
  const int proj = blockIdx.y >> 3, h = blockIdx.y & 7;
  const __bf16* W = (proj == 0 ? Wq : proj == 1 ? Wk : Wv) + (size_t)h * DD * EE;
  const float* bias = (proj == 0 ? bq : proj == 1 ? bk : bv) + h * EE;
  __bf16* OUT = (proj == 0 ? Q : proj == 1 ? K : V);

  // ---- TDM descriptor (invariant parts). 2-D tile: 128 rows x 32 bf16 cols
  // out of a [4096][1024] bf16 tensor; LDS rows padded by 4 DWORDs per 16 DWORDs.
  const unsigned ldsA = (unsigned)(size_t)(LAS*)(&As[0][0]);
  const unsigned long long gbaseA =
      (unsigned long long)(size_t)X + (size_t)m0 * DD * 2;
  u32x8 g1;
  g1[0] = (1u << 16)    // data_size = 2 bytes
        | (1u << 20)    // pad_enable
        | (3u << 22)    // pad_interval: 16 DWORDs
        | (3u << 25);   // pad_amount: 4 DWORDs (= 8 bf16)
  g1[1] = (DD & 0xFFFFu) << 16;                       // tensor_dim0 lo16 (1024)
  g1[2] = (DD >> 16) | (((BB * SS) & 0xFFFFu) << 16); // dim0 hi + dim1 lo (4096)
  g1[3] = ((BB * SS) >> 16) | (32u << 16);            // dim1 hi + tile_dim0 = 32
  g1[4] = 128u;                                       // tile_dim1 = 128
  g1[5] = (unsigned)DD;                               // tensor_dim0_stride = 1024
  g1[6] = 0u;
  g1[7] = 0u;

  v8f acc[4][2] = {};
  for (int kk = 0; kk < DD; kk += 32) {
    if (w == 0) {  // one wave drives the TDM for the A tile
      unsigned long long ga = gbaseA + (unsigned long long)kk * 2u;
      u32x4 g0;
      g0[0] = 1u;                                  // count = 1 valid descriptor
      g0[1] = ldsA;                                // lds_addr
      g0[2] = (unsigned)ga;                        // global_addr lo
      g0[3] = (unsigned)(ga >> 32) | (2u << 30);   // global_addr hi | type=2
      asm volatile("tensor_load_to_lds %0, %1" :: "s"(g0), "s"(g1) : "memory");
      __builtin_amdgcn_s_wait_tensorcnt(0);
    }
    {  // stage B tile 32x128, transposed into Bs[n][k]
      int kr = tid >> 3, nc = (tid & 7) * 16;
      const __bf16* src = W + (size_t)(kk + kr) * EE + nc;
      __builtin_prefetch(src + 32 * EE, 0, 0);
      v16bf t = *(const v16bf*)src;
#pragma unroll
      for (int i = 0; i < 16; i++) Bs[nc + i][kr] = t[i];
    }
    __syncthreads();
    v16bf af[4];
    const int c0 = (lane < 16) ? 0 : 8;
#pragma unroll
    for (int mt = 0; mt < 4; mt++) {
      int row = wm * 64 + mt * 16 + (lane & 15);
      af[mt] = cat8(*(const v8bf*)&As[row][c0], *(const v8bf*)&As[row][c0 + 16]);
    }
    const int k0 = (lane < 16) ? 0 : 8;
#pragma unroll
    for (int nt = 0; nt < 2; nt++) {
      int nrow = wn * 32 + nt * 16 + (lane & 15);
      v16bf bf = cat8(*(const v8bf*)&Bs[nrow][k0], *(const v8bf*)&Bs[nrow][k0 + 16]);
#pragma unroll
      for (int mt = 0; mt < 4; mt++) acc[mt][nt] = wmma_bf16(af[mt], bf, acc[mt][nt]);
    }
    __syncthreads();
  }
  // epilogue: fp32 bias add, store bf16 to [B,H,S,E]
#pragma unroll
  for (int mt = 0; mt < 4; mt++)
#pragma unroll
    for (int nt = 0; nt < 2; nt++) {
      int e = wn * 32 + nt * 16 + (lane & 15);
      float bia = bias[e];
#pragma unroll
      for (int rr = 0; rr < 8; rr++) {
        int m = m0 + wm * 64 + mt * 16 + rr + ((lane >> 4) << 3);
        int b = m >> 11, s = m & (SS - 1);
        OUT[(((size_t)(b * HH + h)) * SS + s) * EE + e] =
            (__bf16)(acc[mt][nt][rr] + bia);
      }
    }
}

// ------------------------------------------------------ banded WMMA attention
// One wave handles 16 decimated queries; keys form a 32-wide strip; per-row
// 16-wide band softmax; padded positions use the projected-zero (bias) row.
__global__ __launch_bounds__(128) void attn_kernel(
    const __bf16* __restrict__ Q, const __bf16* __restrict__ K,
    const __bf16* __restrict__ V, const __bf16* __restrict__ bkb,
    const __bf16* __restrict__ bvb, const int* __restrict__ dil,
    __bf16* __restrict__ ATT) {  // [B][S][H*E]
  __shared__ __align__(16) float  lg[4][16][32];
  __shared__ __align__(16) __bf16 Sc[4][16][32];
  __shared__ __align__(16) __bf16 Vt[4][128][40];  // [e][keypos], +8 pad
  const int tid = threadIdx.x, lane = tid & 31, w = tid >> 5;
  const int h = blockIdx.y, b = blockIdx.z;
  const int d = dil[h];
  const int Sd = SS / d;
  const int tilesPerRes = Sd / 16;
  const int tileIdx = blockIdx.x * 4 + w;  // 0..127
  const int r = tileIdx / tilesPerRes;
  const int t0 = (tileIdx % tilesPerRes) * 16;
  const int shift = (d * (KWW - 1)) / 2;
  int rs = r - shift;
  int r2 = rs % d;
  if (r2 < 0) r2 += d;
  const int off = (rs - r2) / d;  // key strip: pos p = r2 + d*(tbase + jcol)
  const int tbase = t0 + off;

  const size_t headBase = ((size_t)(b * HH + h)) * SS * EE;
  const __bf16* Qh = Q + headBase;
  const __bf16* Kh = K + headBase;
  const __bf16* Vh = V + headBase;
  const __bf16* bkr = bkb + h * EE;
  const __bf16* bvr = bvb + h * EE;

  // Q fragments (A: 16 queries x E, chunked K=32)
  v16bf qf[4];
  {
    int m = lane & 15;
    int s = r + d * (t0 + m);
    const __bf16* qrow = Qh + (size_t)s * EE;
    int c0 = (lane < 16) ? 0 : 16;
#pragma unroll
    for (int kc = 0; kc < 4; kc++)
      qf[kc] = cat8(*(const v8bf*)(qrow + kc * 32 + c0),
                    *(const v8bf*)(qrow + kc * 32 + c0 + 8));
  }
  // logits[16x32] = Q * K^T  (B fragment: lane fixes key position, reads 16 e's)
  v8f lacc[2] = {};
  {
    int k0 = (lane < 16) ? 0 : 16;
#pragma unroll
    for (int nt = 0; nt < 2; nt++) {
      int jcol = nt * 16 + (lane & 15);
      int tp = tbase + jcol;
      const __bf16* krow =
          (tp >= 0 && tp < Sd) ? (Kh + (size_t)(r2 + d * tp) * EE) : bkr;
#pragma unroll
      for (int kc = 0; kc < 4; kc++) {
        v16bf bf = *(const v16bf*)(krow + kc * 32 + k0);
        lacc[nt] = wmma_bf16(qf[kc], bf, lacc[nt]);
      }
    }
  }
  // spill logits, banded softmax (lanes 0-15, one row each)
  {
    int n = lane & 15, mh = (lane >> 4) * 8;
#pragma unroll
    for (int nt = 0; nt < 2; nt++)
#pragma unroll
      for (int rr = 0; rr < 8; rr++) lg[w][mh + rr][nt * 16 + n] = lacc[nt][rr];
  }
  // stage V strip transposed: Vt[e][keypos]
  {
    int j = lane;  // key strip column 0..31
    int tp = tbase + j;
    const __bf16* vrow =
        (tp >= 0 && tp < Sd) ? (Vh + (size_t)(r2 + d * tp) * EE) : bvr;
#pragma unroll
    for (int c = 0; c < 8; c++) {
      v16bf vv = *(const v16bf*)(vrow + c * 16);
#pragma unroll
      for (int i = 0; i < 16; i++) Vt[w][c * 16 + i][j] = vv[i];
    }
  }
  __syncthreads();
  if (lane < 16) {
    int m = lane;
    float vals[16], vmax = -3.4e38f;
#pragma unroll
    for (int j = 0; j < 16; j++) {
      vals[j] = lg[w][m][m + j];  // band: row m uses strip cols m..m+15
      vmax = fmaxf(vmax, vals[j]);
    }
    float ssum = 0.f;
#pragma unroll
    for (int j = 0; j < 16; j++) {
      vals[j] = __expf(vals[j] - vmax);
      ssum += vals[j];
    }
    float inv = 1.f / ssum;
#pragma unroll
    for (int j = 0; j < 32; j++) Sc[w][m][j] = (__bf16)0.f;
#pragma unroll
    for (int j = 0; j < 16; j++) Sc[w][m][m + j] = (__bf16)(vals[j] * inv);
  }
  __syncthreads();
  // attn[16x128] = Sc(16x32) * Vstrip(32x128), then /sqrt(E) (reference quirk)
  v16bf scf;
  {
    int m = lane & 15, c0 = (lane < 16) ? 0 : 8;
    scf = cat8(*(const v8bf*)&Sc[w][m][c0], *(const v8bf*)&Sc[w][m][c0 + 16]);
  }
  const float scale = 0.0883883476483184f;  // 1/sqrt(128)
  const int k0 = (lane < 16) ? 0 : 8;
  const int n = lane & 15, mh = (lane >> 4) * 8;
#pragma unroll
  for (int nt = 0; nt < 8; nt++) {
    int e = nt * 16 + n;
    v16bf bf = cat8(*(const v8bf*)&Vt[w][e][k0], *(const v8bf*)&Vt[w][e][k0 + 16]);
    v8f z = {};
    v8f o = wmma_bf16(scf, bf, z);
#pragma unroll
    for (int rr = 0; rr < 8; rr++) {
      int m = mh + rr;
      int s = r + d * (t0 + m);
      ATT[((size_t)(b * SS + s)) * (HH * EE) + h * EE + e] = (__bf16)(o[rr] * scale);
    }
  }
}

// ------------------------------------------------------------ output GEMM
// out[4096x1024] = attn[4096x1024] * Wo[1024x1024] + bo, fp32 out.
// A tile staged with async global->LDS copies when the builtins exist.
__global__ __launch_bounds__(256) void out_gemm(const __bf16* __restrict__ A,
                                                const __bf16* __restrict__ Wo,
                                                const float* __restrict__ bo,
                                                float* __restrict__ OUT) {
  __shared__ __align__(16) __bf16 As[128][40];
  __shared__ __align__(16) __bf16 Bs[128][40];
  const int tid = threadIdx.x, lane = tid & 31, w = tid >> 5;
  const int wm = w & 1, wn = w >> 1;
  const int m0 = blockIdx.x * 128, n0 = blockIdx.y * 128;
  v8f acc[4][2] = {};
  for (int kk = 0; kk < HH * EE; kk += 32) {
    {
      int row = tid >> 1, ch = (tid & 1) * 16;
      const __bf16* src = A + (size_t)(m0 + row) * (HH * EE) + kk + ch;
#if HAS_ASYNC_LDS
      __builtin_amdgcn_global_load_async_to_lds_b128(
          (GV4i*)src, (LV4i*)&As[row][ch], 0, 0);
      __builtin_amdgcn_global_load_async_to_lds_b128(
          (GV4i*)(src + 8), (LV4i*)&As[row][ch + 8], 0, 0);
#else
      __builtin_prefetch(src + 32, 0, 0);
      *(v8bf*)&As[row][ch]     = *(const v8bf*)src;
      *(v8bf*)&As[row][ch + 8] = *(const v8bf*)(src + 8);
#endif
    }
    {
      int kr = tid >> 3, nc = (tid & 7) * 16;
      const __bf16* src = Wo + (size_t)(kk + kr) * DD + n0 + nc;
      __builtin_prefetch(src + 32 * DD, 0, 0);
      v16bf t = *(const v16bf*)src;
#pragma unroll
      for (int i = 0; i < 16; i++) Bs[nc + i][kr] = t[i];
    }
#if HAS_ASYNC_LDS
    __builtin_amdgcn_s_wait_asynccnt(0);
#endif
    __syncthreads();
    v16bf af[4];
    const int c0 = (lane < 16) ? 0 : 8;
#pragma unroll
    for (int mt = 0; mt < 4; mt++) {
      int row = wm * 64 + mt * 16 + (lane & 15);
      af[mt] = cat8(*(const v8bf*)&As[row][c0], *(const v8bf*)&As[row][c0 + 16]);
    }
    const int k0 = (lane < 16) ? 0 : 8;
#pragma unroll
    for (int nt = 0; nt < 2; nt++) {
      int nrow = wn * 32 + nt * 16 + (lane & 15);
      v16bf bf = cat8(*(const v8bf*)&Bs[nrow][k0], *(const v8bf*)&Bs[nrow][k0 + 16]);
#pragma unroll
      for (int mt = 0; mt < 4; mt++) acc[mt][nt] = wmma_bf16(af[mt], bf, acc[mt][nt]);
    }
    __syncthreads();
  }
#pragma unroll
  for (int mt = 0; mt < 4; mt++)
#pragma unroll
    for (int nt = 0; nt < 2; nt++) {
      int nn = n0 + wn * 32 + nt * 16 + (lane & 15);
      float bia = bo[nn];
#pragma unroll
      for (int rr = 0; rr < 8; rr++) {
        int m = m0 + wm * 64 + mt * 16 + rr + ((lane >> 4) << 3);
        OUT[(size_t)m * DD + nn] = acc[mt][nt][rr] + bia;
      }
    }
}

// ----------------------------------------------------------------- launcher
extern "C" void kernel_launch(void* const* d_in, const int* in_sizes, int n_in,
                              void* d_out, int out_size, void* d_ws,
                              size_t ws_size, hipStream_t stream) {
  const float* x  = (const float*)d_in[0];
  const float* Wq = (const float*)d_in[1];
  const float* bq = (const float*)d_in[2];
  const float* Wk = (const float*)d_in[3];
  const float* bk = (const float*)d_in[4];
  const float* Wv = (const float*)d_in[5];
  const float* bv = (const float*)d_in[6];
  const float* Wo = (const float*)d_in[7];
  const float* bo = (const float*)d_in[8];
  const int* dil  = (const int*)d_in[9];
  float* out = (float*)d_out;

  __bf16* ws = (__bf16*)d_ws;
  size_t o = 0;
  __bf16* xb  = ws + o; o += (size_t)BB * SS * DD;   // x bf16
  __bf16* Wqb = ws + o; o += (size_t)HH * DD * EE;
  __bf16* Wkb = ws + o; o += (size_t)HH * DD * EE;
  __bf16* Wvb = ws + o; o += (size_t)HH * DD * EE;
  __bf16* Wob = ws + o; o += (size_t)HH * EE * DD;
  __bf16* qb  = ws + o; o += (size_t)BB * HH * SS * EE;
  __bf16* kb  = ws + o; o += (size_t)BB * HH * SS * EE;
  __bf16* vb  = ws + o; o += (size_t)BB * HH * SS * EE;
  __bf16* atb = ws + o; o += (size_t)BB * SS * HH * EE;
  __bf16* bkb = ws + o; o += (size_t)HH * EE;
  __bf16* bvb = ws + o; o += (size_t)HH * EE;

  const int nx = BB * SS * DD, nw = HH * DD * EE, nb = HH * EE;
  cvt_f32_bf16<<<(nx + 255) / 256, 256, 0, stream>>>(x, xb, nx);
  cvt_f32_bf16<<<(nw + 255) / 256, 256, 0, stream>>>(Wq, Wqb, nw);
  cvt_f32_bf16<<<(nw + 255) / 256, 256, 0, stream>>>(Wk, Wkb, nw);
  cvt_f32_bf16<<<(nw + 255) / 256, 256, 0, stream>>>(Wv, Wvb, nw);
  cvt_f32_bf16<<<(nw + 255) / 256, 256, 0, stream>>>(Wo, Wob, nw);
  cvt_f32_bf16<<<(nb + 255) / 256, 256, 0, stream>>>(bk, bkb, nb);
  cvt_f32_bf16<<<(nb + 255) / 256, 256, 0, stream>>>(bv, bvb, nb);

  qkv_gemm<<<dim3(BB * SS / 128, 3 * HH), 256, 0, stream>>>(
      xb, Wqb, Wkb, Wvb, bq, bk, bv, qb, kb, vb);

  attn_kernel<<<dim3(SS / 16 / 4, HH, BB), 128, 0, stream>>>(qb, kb, vb, bkb,
                                                             bvb, dil, atb);

  out_gemm<<<dim3(BB * SS / 128, DD / 128), 256, 0, stream>>>(atb, Wob, bo, out);
}
